// GraphMambaPhysicsModel_24584392802966
// MI455X (gfx1250) — matrix operand order repaired and checked by reading, not compiled, not run on hardware
//
#include <hip/hip_runtime.h>
#include <hip/hip_bf16.h>

typedef __bf16 bf16;
typedef __attribute__((ext_vector_type(16))) __bf16 v16bf;
typedef __attribute__((ext_vector_type(8)))  float  v8f;

#define GG 2048   // B*T graphs
#define NN 118    // nodes
#define EE 372    // edges
#define DD 128    // hidden
#define TT 256    // seq len
#define BB 8      // batch

struct alignas(16) Q16 { unsigned int x, y, z, w; };
struct Frag32 { Q16 lo, hi; };

__device__ __forceinline__ v16bf ld_frag(const bf16* p0, const bf16* p1) {
    Frag32 f;
    f.lo = *reinterpret_cast<const Q16*>(p0);
    f.hi = *reinterpret_cast<const Q16*>(p1);
    return __builtin_bit_cast(v16bf, f);
}
__device__ __forceinline__ v16bf zero_frag() {
    Frag32 f{};
    return __builtin_bit_cast(v16bf, f);
}
__device__ __forceinline__ float sigf(float x)  { return 1.f / (1.f + __expf(-x)); }
__device__ __forceinline__ float siluf(float x) { return x   / (1.f + __expf(-x)); }

// ---------------------------------------------------------------- prep: degree / norm
__global__ void prep_deg(const long long* __restrict__ edges,
                         float* __restrict__ invdegG, float* __restrict__ normG) {
    __shared__ float deg[NN];
    __shared__ float isq[NN];
    int tid = threadIdx.x;
    if (tid < NN) deg[tid] = 1.f;
    __syncthreads();
    for (int e = tid; e < EE; e += blockDim.x)
        atomicAdd(&deg[(int)edges[EE + e]], 1.f);          // exact integer adds -> deterministic
    __syncthreads();
    if (tid < NN) { float d = deg[tid]; invdegG[tid] = 1.f / d; isq[tid] = rsqrtf(d); }
    __syncthreads();
    for (int e = tid; e < EE; e += blockDim.x)
        normG[e] = isq[(int)edges[e]] * isq[(int)edges[EE + e]];
}

// ---------------------------------------------------------------- prep: weight conversion
// Wt[n*128+k] = bf16(W[k*128+n])  (n-major so B-fragments are K-contiguous)
// WcT[layer][n*256+k] = bf16(k<128 ? Wih[n][k] : Whh[n][k-128])
__global__ void prep_weights(const float* __restrict__ W2, const float* __restrict__ W3,
                             const float* __restrict__ Wih0, const float* __restrict__ Whh0,
                             const float* __restrict__ Wih1, const float* __restrict__ Whh1,
                             bf16* __restrict__ Wt2, bf16* __restrict__ Wt3,
                             bf16* __restrict__ WcT) {
    int i = blockIdx.x * blockDim.x + threadIdx.x;
    if (i < 16384) {
        int k = i >> 7, n = i & 127;
        Wt2[n * DD + k] = (bf16)W2[i];
        Wt3[n * DD + k] = (bf16)W3[i];
    }
    if (i < 262144) {                       // 2 layers * 512 * 256
        int l = i >> 17;
        int rem = i & 131071;
        int n = rem >> 8, k = rem & 255;
        const float* Wih = l ? Wih1 : Wih0;
        const float* Whh = l ? Whh1 : Whh0;
        float v = (k < 128) ? Wih[n * 128 + k] : Whh[n * 128 + (k - 128)];
        WcT[i] = (bf16)v;
    }
}

// ---------------------------------------------------------------- GCN layer 1 (K=3, VALU)
__global__ void __launch_bounds__(128)
gcn_first(const float* __restrict__ snap, const float* __restrict__ scale,
          const float* __restrict__ shift, const float* __restrict__ W1,
          const float* __restrict__ b1, const long long* __restrict__ edges,
          const float* __restrict__ invdegG, const float* __restrict__ normG,
          bf16* __restrict__ actOut) {
    extern __shared__ char smem[];
    float* h    = (float*)smem;            // 118*128
    float* agg  = h   + NN * DD;           // 118*128
    float* xs   = agg + NN * DD;           // 118*3
    float* eN   = xs  + NN * 3;            // 372
    float* invd = eN  + EE;                // 118
    int*   eS   = (int*)(invd + NN);       // 372
    int*   eD   = eS + EE;                 // 372

    const int tid = threadIdx.x;
    const int g   = blockIdx.x;

    for (int i = tid; i < NN * 3; i += 128) {
        int f = i % 3;
        xs[i] = snap[(size_t)g * NN * 3 + i] * scale[f] + shift[f];
    }
    for (int i = tid; i < EE; i += 128) {
        eS[i] = (int)edges[i]; eD[i] = (int)edges[EE + i]; eN[i] = normG[i];
    }
    for (int i = tid; i < NN; i += 128) invd[i] = invdegG[i];
    __syncthreads();

    const int c = tid;                      // thread owns feature column c (no cross-thread hazards)
    const float w0 = W1[c], w1 = W1[DD + c], w2 = W1[2 * DD + c];
    for (int n = 0; n < NN; ++n)
        h[n * DD + c] = xs[n * 3] * w0 + xs[n * 3 + 1] * w1 + xs[n * 3 + 2] * w2;
    for (int n = 0; n < NN; ++n) agg[n * DD + c] = h[n * DD + c] * invd[n];
    for (int e = 0; e < EE; ++e) agg[eD[e] * DD + c] += h[eS[e] * DD + c] * eN[e];
    const float bc = b1[c];
    for (int n = 0; n < NN; ++n) {
        float v = agg[n * DD + c] + bc;
        actOut[((size_t)g * NN + n) * DD + c] = (bf16)siluf(v);
    }
}

// ---------------------------------------------------------------- GCN layers 2/3 (WMMA)
__global__ void __launch_bounds__(256)
gcn_wmma(const bf16* __restrict__ actIn, const bf16* __restrict__ Wt,
         const float* __restrict__ bias, const long long* __restrict__ edges,
         const float* __restrict__ invdegG, const float* __restrict__ normG,
         bf16* __restrict__ actOut, float* __restrict__ emb) {
    extern __shared__ char smem[];
    float* h    = (float*)smem;            // 128*128 (rows 118..127 never read)
    float* agg  = h   + 128 * DD;          // 118*128
    float* eN   = agg + NN * DD;           // 372
    float* invd = eN  + EE;                // 118
    int*   eS   = (int*)(invd + NN);       // 372
    int*   eD   = eS + EE;                 // 372

    const int tid = threadIdx.x;
    const int g   = blockIdx.x;

    for (int i = tid; i < EE; i += 256) {
        eS[i] = (int)edges[i]; eD[i] = (int)edges[EE + i]; eN[i] = normG[i];
    }
    for (int i = tid; i < NN; i += 256) invd[i] = invdegG[i];

    // ---- h = actIn(g) [118(->128) x 128] @ W [128 x 128] via v_wmma_f32_16x16x32_bf16
    const int wv   = tid >> 5;             // wave = M-tile (8 waves x 8 N-tiles)
    const int lane = tid & 31;
    const int l15  = lane & 15;
    const int hi   = lane >> 4;

    v8f acc[8];
    #pragma unroll
    for (int nt = 0; nt < 8; ++nt)
        #pragma unroll
        for (int r = 0; r < 8; ++r) acc[nt][r] = 0.f;

    const int m = wv * 16 + l15;
    const size_t rowbase = ((size_t)g * NN + m) * DD;

    #pragma unroll
    for (int kc = 0; kc < 4; ++kc) {
        v16bf av;
        if (m < NN) {
            const bf16* ap = actIn + rowbase + kc * 32 + hi * 8;
            av = ld_frag(ap, ap + 16);
        } else {
            av = zero_frag();              // pad rows: EXEC stays all-ones for WMMA
        }
        #pragma unroll
        for (int nt = 0; nt < 8; ++nt) {
            const bf16* bp = Wt + (nt * 16 + l15) * DD + kc * 32 + hi * 8;
            v16bf bv = ld_frag(bp, bp + 16);
            acc[nt] = __builtin_amdgcn_wmma_f32_16x16x32_bf16(
                false, av, false, bv, (short)0, acc[nt], false, false);
        }
    }
    #pragma unroll
    for (int nt = 0; nt < 8; ++nt)
        #pragma unroll
        for (int r = 0; r < 8; ++r)
            h[(wv * 16 + hi * 8 + r) * DD + nt * 16 + l15] = acc[nt][r];

    __syncthreads();

    // ---- aggregation + bias + SiLU (+ node mean on last layer), column-owned
    if (tid < DD) {
        const int c = tid;
        for (int n = 0; n < NN; ++n) agg[n * DD + c] = h[n * DD + c] * invd[n];
        for (int e = 0; e < EE; ++e) agg[eD[e] * DD + c] += h[eS[e] * DD + c] * eN[e];
        const float bc = bias[c];
        float s = 0.f;
        for (int n = 0; n < NN; ++n) {
            float v  = agg[n * DD + c] + bc;
            float sv = siluf(v);
            actOut[((size_t)g * NN + n) * DD + c] = (bf16)sv;
            s += sv;
        }
        if (emb) emb[(size_t)g * DD + c] = s * (1.f / (float)NN);
    }
}

// ---------------------------------------------------------------- 2-layer LSTM (persistent, WMMA)
__global__ void __launch_bounds__(512)
lstm2(const float* __restrict__ emb, const bf16* __restrict__ WcT,
      const float* __restrict__ bih0, const float* __restrict__ bhh0,
      const float* __restrict__ bih1, const float* __restrict__ bhh1,
      float* __restrict__ hseq, float* __restrict__ finalh) {
    extern __shared__ char smem[];
    bf16*  xh    = (bf16*)smem;                         // 16 x 256 (rows 8..15 zero)
    float* gates = (float*)(smem + 8192);               // 8 x 512
    float* cst   = (float*)(smem + 8192 + 16384);       // 8 x 128
    float* bsum  = (float*)(smem + 8192 + 16384 + 4096);// 512

    const int tid  = threadIdx.x;
    const int wv   = tid >> 5;     // 16 waves, 2 N-tiles each -> 32 gate cols per wave
    const int lane = tid & 31;
    const int l15  = lane & 15;
    const int hi   = lane >> 4;

    for (int layer = 0; layer < 2; ++layer) {
        const bf16*  Wc = WcT + (size_t)layer * 512 * 256;
        const float* bi = layer ? bih1 : bih0;
        const float* bh = layer ? bhh1 : bhh0;
        for (int i = tid; i < 512; i += 512)       bsum[i] = bi[i] + bh[i];
        for (int i = tid; i < BB * DD; i += 512)   cst[i]  = 0.f;
        for (int i = tid; i < 16 * 256; i += 512)  xh[i]   = (bf16)0.f;   // h0=0, pad rows=0
        __syncthreads();

        const float* xsrc = layer ? hseq : emb;
        for (int t = 0; t < TT; ++t) {
            // stage x_t (bf16) into xh[:, 0:128]
            for (int i = tid; i < BB * DD; i += 512) {
                int b = i >> 7, d = i & 127;
                xh[b * 256 + d] = (bf16)xsrc[((size_t)b * TT + t) * DD + d];
            }
            __syncthreads();

            // gates(8x512) = [x,h](8x256) @ WcT^T : M=16(pad), N=512, K=256
            v8f accA, accB;
            #pragma unroll
            for (int r = 0; r < 8; ++r) { accA[r] = 0.f; accB[r] = 0.f; }
            const int n0 = wv * 32 + l15;
            #pragma unroll
            for (int kc = 0; kc < 8; ++kc) {
                const bf16* ap = xh + l15 * 256 + kc * 32 + hi * 8;
                v16bf av = ld_frag(ap, ap + 16);
                const bf16* bp0 = Wc + (size_t)n0 * 256 + kc * 32 + hi * 8;
                v16bf bv0 = ld_frag(bp0, bp0 + 16);
                const bf16* bp1 = bp0 + 16 * 256;
                v16bf bv1 = ld_frag(bp1, bp1 + 16);
                accA = __builtin_amdgcn_wmma_f32_16x16x32_bf16(
                    false, av, false, bv0, (short)0, accA, false, false);
                accB = __builtin_amdgcn_wmma_f32_16x16x32_bf16(
                    false, av, false, bv1, (short)0, accB, false, false);
            }
            if (lane < 16) {                 // lanes 0..15 hold M=r (batch rows 0..7)
                #pragma unroll
                for (int r = 0; r < 8; ++r) {
                    gates[r * 512 + wv * 32 + l15]      = accA[r];
                    gates[r * 512 + wv * 32 + 16 + l15] = accB[r];
                }
            }
            __syncthreads();

            // elementwise LSTM cell
            for (int i = tid; i < BB * DD; i += 512) {
                int b = i >> 7, d = i & 127;
                float ig = gates[b * 512 +       d] + bsum[d];
                float fg = gates[b * 512 + 128 + d] + bsum[128 + d];
                float gg = gates[b * 512 + 256 + d] + bsum[256 + d];
                float og = gates[b * 512 + 384 + d] + bsum[384 + d];
                float cn = sigf(fg) * cst[b * DD + d] + sigf(ig) * tanhf(gg);
                float hn = sigf(og) * tanhf(cn);
                cst[b * DD + d]   = cn;
                xh[b * 256 + 128 + d] = (bf16)hn;     // h feeds next step
                if (layer == 0)            hseq[((size_t)b * TT + t) * DD + d] = hn;
                else if (t == TT - 1)      finalh[b * DD + d] = hn;
            }
            __syncthreads();
        }
        __threadfence();      // make hseq globally visible before layer 1 reads it
        __syncthreads();
    }
}

// ---------------------------------------------------------------- head MLP + softplus
__global__ void __launch_bounds__(256)
head_mlp(const float* __restrict__ finalh,
         const float* __restrict__ hW1, const float* __restrict__ hb1,
         const float* __restrict__ hW2, const float* __restrict__ hb2,
         const float* __restrict__ hW3, const float* __restrict__ hb3,
         float* __restrict__ out) {
    extern __shared__ char smem[];
    float* fin = (float*)smem;      // 8*128
    float* y1  = fin + 1024;        // 8*128
    float* y2  = y1 + 1024;         // 8*64
    int tid = threadIdx.x;
    for (int i = tid; i < 1024; i += 256) fin[i] = finalh[i];
    __syncthreads();
    for (int i = tid; i < 1024; i += 256) {
        int b = i >> 7, j = i & 127;
        float s = hb1[j];
        for (int k = 0; k < 128; ++k) s += fin[b * 128 + k] * hW1[k * 128 + j];
        y1[i] = siluf(s);
    }
    __syncthreads();
    for (int i = tid; i < 512; i += 256) {
        int b = i >> 6, j = i & 63;
        float s = hb2[j];
        for (int k = 0; k < 128; ++k) s += y1[b * 128 + k] * hW2[k * 64 + j];
        y2[i] = siluf(s);
    }
    __syncthreads();
    if (tid < 16) {
        int b = tid >> 1, j = tid & 1;
        float s = hb3[j];
        for (int k = 0; k < 64; ++k) s += y2[b * 64 + k] * hW3[k * 2 + j];
        float sp = (s > 20.f) ? s : log1pf(__expf(s));
        out[b * 2 + j] = sp + 1e-6f;
    }
}

// ---------------------------------------------------------------- launch
extern "C" void kernel_launch(void* const* d_in, const int* in_sizes, int n_in,
                              void* d_out, int out_size, void* d_ws, size_t ws_size,
                              hipStream_t stream) {
    (void)in_sizes; (void)n_in; (void)out_size; (void)ws_size;
    const float*     snap  = (const float*)d_in[0];
    const long long* edges = (const long long*)d_in[1];
    const float* scale = (const float*)d_in[2];
    const float* shift = (const float*)d_in[3];
    const float* W1 = (const float*)d_in[4];
    const float* b1 = (const float*)d_in[5];
    const float* W2 = (const float*)d_in[6];
    const float* b2 = (const float*)d_in[7];
    const float* W3 = (const float*)d_in[8];
    const float* b3 = (const float*)d_in[9];
    const float* Wih0 = (const float*)d_in[10];
    const float* Whh0 = (const float*)d_in[11];
    const float* bih0 = (const float*)d_in[12];
    const float* bhh0 = (const float*)d_in[13];
    const float* Wih1 = (const float*)d_in[14];
    const float* Whh1 = (const float*)d_in[15];
    const float* bih1 = (const float*)d_in[16];
    const float* bhh1 = (const float*)d_in[17];
    const float* hW1 = (const float*)d_in[18];
    const float* hb1 = (const float*)d_in[19];
    const float* hW2 = (const float*)d_in[20];
    const float* hb2 = (const float*)d_in[21];
    const float* hW3 = (const float*)d_in[22];
    const float* hb3 = (const float*)d_in[23];
    float* out = (float*)d_out;

    char* w = (char*)d_ws;
    auto take = [&](size_t bytes) -> char* {
        char* p = w;
        w += (bytes + 255) & ~(size_t)255;
        return p;
    };
    bf16*  actA   = (bf16*)take((size_t)GG * NN * DD * sizeof(bf16));  // ~59 MB
    bf16*  actB   = (bf16*)take((size_t)GG * NN * DD * sizeof(bf16));  // ~59 MB
    bf16*  Wt2    = (bf16*)take(16384 * sizeof(bf16));
    bf16*  Wt3    = (bf16*)take(16384 * sizeof(bf16));
    bf16*  WcT    = (bf16*)take(262144 * sizeof(bf16));
    float* invdeg = (float*)take(NN * sizeof(float));
    float* normv  = (float*)take(EE * sizeof(float));
    float* emb    = (float*)take((size_t)GG * DD * sizeof(float));
    float* hseq   = (float*)take((size_t)BB * TT * DD * sizeof(float));
    float* finalh = (float*)take(BB * DD * sizeof(float));

    prep_deg<<<1, 128, 0, stream>>>(edges, invdeg, normv);
    prep_weights<<<1024, 256, 0, stream>>>(W2, W3, Wih0, Whh0, Wih1, Whh1, Wt2, Wt3, WcT);

    size_t lds1 = (size_t)(NN * DD + NN * DD + NN * 3 + EE + NN) * 4 + (size_t)EE * 4 * 2; // ~127 KB
    gcn_first<<<GG, 128, lds1, stream>>>(snap, scale, shift, W1, b1, edges, invdeg, normv, actA);

    size_t lds2 = (size_t)(128 * DD + NN * DD + EE + NN) * 4 + (size_t)EE * 4 * 2;         // ~131 KB
    gcn_wmma<<<GG, 256, lds2, stream>>>(actA, Wt2, b2, edges, invdeg, normv, actB, nullptr);
    gcn_wmma<<<GG, 256, lds2, stream>>>(actB, Wt3, b3, edges, invdeg, normv, actA, emb);

    lstm2<<<1, 512, 8192 + 16384 + 4096 + 2048, stream>>>(emb, WcT, bih0, bhh0, bih1, bhh1,
                                                          hseq, finalh);

    head_mlp<<<1, 256, 4096 + 4096 + 2048, stream>>>(finalh, hW1, hb1, hW2, hb2, hW3, hb3, out);
}
